// Qwen2MoeSharedExpert_53042846105777
// MI455X (gfx1250) — matrix operand order repaired and checked by reading, not compiled, not run on hardware
//
#include <hip/hip_runtime.h>
#include <hip/hip_bf16.h>

typedef __attribute__((ext_vector_type(16))) _Float16 v16h;
typedef __attribute__((ext_vector_type(8)))  _Float16 v8h;
typedef __attribute__((ext_vector_type(4)))  _Float16 v4h;
typedef __attribute__((ext_vector_type(8)))  float    v8f;
typedef __attribute__((ext_vector_type(4)))  unsigned int u32x4;
typedef __attribute__((ext_vector_type(8)))  int      i32x8;
typedef __attribute__((ext_vector_type(4)))  int      i32x4;

union V16 { v16h v; v8h h[2]; };

// Problem sizes (fixed by the reference)
constexpr int Tt  = 4096;
constexpr int Hh  = 2048;
constexpr int Ii  = 5632;

// Tiling
constexpr int BM  = 128;   // rows per block
constexpr int BN  = 128;   // output columns per block
constexpr int BK  = 32;    // K step == WMMA K
constexpr int LDA = 40;    // padded LDS row stride (halves) = 80B, 16B-aligned
constexpr int LDB = 40;

// ---------------------------------------------------------------------------
// Kernel 1: h[T,I] (f16) = silu(x @ dequant(gate)) * (x @ dequant(up))
// Waves: 2(M) x 4(N); per-wave tile 64x32 => 4x2 WMMA tiles per weight matrix.
// ---------------------------------------------------------------------------
__global__ __launch_bounds__(256)
void gateup_swiglu_kernel(const float* __restrict__ x,
                          const int*   __restrict__ gq,
                          const float* __restrict__ gs,
                          const int*   __restrict__ gz,
                          const int*   __restrict__ uq,
                          const float* __restrict__ us,
                          const int*   __restrict__ uz,
                          _Float16*    __restrict__ hbuf)
{
    __shared__ __align__(16) _Float16 As [BM * LDA];  // [m][k]
    __shared__ __align__(16) _Float16 Bgs[BN * LDB];  // [n][k] (K-contiguous)
    __shared__ __align__(16) _Float16 Bus[BN * LDB];

    const int tid   = threadIdx.x;
    const int iBase = blockIdx.x * BN;
    const int tBase = blockIdx.y * BM;

    const int wave  = tid >> 5;
    const int lane  = tid & 31;
    const int wm    = wave >> 2;      // 0..1
    const int wn    = wave & 3;       // 0..3
    const int m0    = wm * 64;
    const int n0    = wn * 32;
    const int lhalf = lane >> 4;      // 0 or 1
    const int l16   = lane & 15;

    v8f accg[4][2] = {};
    v8f accu[4][2] = {};

    // B staging map: column bn within tile, base k quad bk0
    const int bn  = tid & 127;
    const int bk0 = (tid >> 7) * 4;   // 0 or 4

    for (int k0 = 0; k0 < Hh; k0 += BK) {
        // ---------------- stage A: 128x32 f32 -> f16 ----------------
        #pragma unroll
        for (int cc = 0; cc < 4; ++cc) {
            const int c  = tid + cc * 256;        // 1024 chunks of 4 floats
            const int r  = c >> 3;
            const int kc = (c & 7) * 4;
            const float4 f = *reinterpret_cast<const float4*>(
                x + (size_t)(tBase + r) * Hh + k0 + kc);
            v4h h4 = { (_Float16)f.x, (_Float16)f.y, (_Float16)f.z, (_Float16)f.w };
            *reinterpret_cast<v4h*>(&As[r * LDA + kc]) = h4;
        }

        // ---------------- stage B: dequant int4 -> f16 --------------
        const int   g   = k0 >> 7;  // GROUP = 128, constant across the 32-tile
        const float sgf = gs[(size_t)g * Ii + iBase + bn];
        const int   zgv = gz[(size_t)g * Ii + iBase + bn];
        const float suf = us[(size_t)g * Ii + iBase + bn];
        const int   zuv = uz[(size_t)g * Ii + iBase + bn];
        #pragma unroll
        for (int j = 0; j < 4; ++j) {
            const int kr = bk0 + j * 8;           // 4 consecutive k rows
            v4h hg, hu;
            #pragma unroll
            for (int e = 0; e < 4; ++e) {
                const size_t off = (size_t)(k0 + kr + e) * Ii + iBase + bn;
                hg[e] = (_Float16)((float)(gq[off] - zgv) * sgf);
                hu[e] = (_Float16)((float)(uq[off] - zuv) * suf);
            }
            *reinterpret_cast<v4h*>(&Bgs[bn * LDB + kr]) = hg;
            *reinterpret_cast<v4h*>(&Bus[bn * LDB + kr]) = hu;
        }

        // prefetch next K tile (global_prefetch_b8)
        if (k0 + BK < Hh) {
            __builtin_prefetch(x  + (size_t)(tBase + (tid >> 1)) * Hh + k0 + BK, 0, 1);
            __builtin_prefetch(gq + (size_t)(k0 + BK + bk0) * Ii + iBase + bn, 0, 1);
            __builtin_prefetch(uq + (size_t)(k0 + BK + bk0) * Ii + iBase + bn, 0, 1);
        }

        __syncthreads();

        // ---------------- WMMA compute ------------------------------
        V16 a[4], bg[2], bu[2];
        #pragma unroll
        for (int mi = 0; mi < 4; ++mi) {
            const int m    = m0 + mi * 16 + l16;
            const int koff = lhalf * 8;
            a[mi].h[0] = *reinterpret_cast<const v8h*>(&As[m * LDA + koff]);
            a[mi].h[1] = *reinterpret_cast<const v8h*>(&As[m * LDA + 16 + koff]);
        }
        #pragma unroll
        for (int ni = 0; ni < 2; ++ni) {
            const int n  = n0 + ni * 16 + l16;
            const int kb = lhalf * 8;
            bg[ni].h[0] = *reinterpret_cast<const v8h*>(&Bgs[n * LDB + kb]);
            bg[ni].h[1] = *reinterpret_cast<const v8h*>(&Bgs[n * LDB + kb + 16]);
            bu[ni].h[0] = *reinterpret_cast<const v8h*>(&Bus[n * LDB + kb]);
            bu[ni].h[1] = *reinterpret_cast<const v8h*>(&Bus[n * LDB + kb + 16]);
        }
        #pragma unroll
        for (int mi = 0; mi < 4; ++mi) {
            #pragma unroll
            for (int ni = 0; ni < 2; ++ni) {
                accg[mi][ni] = __builtin_amdgcn_wmma_f32_16x16x32_f16(
                    false, a[mi].v, false, bg[ni].v, (short)0, accg[mi][ni], false, false);
                accu[mi][ni] = __builtin_amdgcn_wmma_f32_16x16x32_f16(
                    false, a[mi].v, false, bu[ni].v, (short)0, accu[mi][ni], false, false);
            }
        }
        __syncthreads();
    }

    // ---------------- epilogue: SwiGLU, store f16 -------------------
    #pragma unroll
    for (int mi = 0; mi < 4; ++mi) {
        #pragma unroll
        for (int ni = 0; ni < 2; ++ni) {
            #pragma unroll
            for (int r = 0; r < 8; ++r) {
                const int row = tBase + m0 + mi * 16 + lhalf * 8 + r;
                const int col = iBase + n0 + ni * 16 + l16;
                const float gv = accg[mi][ni][r];
                const float uv = accu[mi][ni][r];
                const float hv = gv / (1.0f + __expf(-gv)) * uv;   // silu(g)*u
                hbuf[(size_t)row * Ii + col] = (_Float16)hv;
            }
        }
    }
}

// ---------------------------------------------------------------------------
// Kernel 2: out[T,H] (f32) = h @ dequant(down)
// A tile (f16 pass-through) staged via Tensor Data Mover when available.
// ---------------------------------------------------------------------------
__global__ __launch_bounds__(256)
void down_kernel(const _Float16* __restrict__ hbuf,
                 const int*      __restrict__ dq,
                 const float*    __restrict__ dsc,
                 const int*      __restrict__ dz,
                 float*          __restrict__ out)
{
    __shared__ __align__(16) _Float16 As[BM * LDA];
    __shared__ __align__(16) _Float16 Bs[BN * LDB];

    const int tid   = threadIdx.x;
    const int hBase = blockIdx.x * BN;
    const int tBase = blockIdx.y * BM;

    const int wave  = tid >> 5;
    const int lane  = tid & 31;
    const int wm    = wave >> 2;
    const int wn    = wave & 3;
    const int m0    = wm * 64;
    const int n0    = wn * 32;
    const int lhalf = lane >> 4;
    const int l16   = lane & 15;

    v8f acc[4][2] = {};

    const int bn  = tid & 127;
    const int bk0 = (tid >> 7) * 4;

#if __has_builtin(__builtin_amdgcn_tensor_load_to_lds)
    // LDS byte offset of As for the TDM descriptor
    const unsigned int ldsA =
        (unsigned int)(uintptr_t)(__attribute__((address_space(3))) _Float16*)As;
#endif

    for (int k0 = 0; k0 < Ii; k0 += BK) {
        // ---------------- stage A: TDM tile load --------------------
#if __has_builtin(__builtin_amdgcn_tensor_load_to_lds)
        if (wave == 0) {
            const unsigned long long ga =
                (unsigned long long)(uintptr_t)(hbuf + (size_t)tBase * Ii + k0);
            u32x4 g0;
            g0[0] = 1u;                                   // count=1, user mode
            g0[1] = ldsA;                                 // lds_addr (bytes)
            g0[2] = (unsigned int)ga;                     // global_addr[31:0]
            g0[3] = (unsigned int)(ga >> 32) | (2u << 30);// addr[56:32] | type=2
            i32x8 g1;
            // data_size=2B (1<<16), pad_enable (1<<20),
            // pad_interval=16 DWORDs (3<<22), pad_amount=4 DWORDs (3<<25)
            g1[0] = (int)((1u << 16) | (1u << 20) | (3u << 22) | (3u << 25));
            g1[1] = (int)((unsigned int)Ii << 16);        // tensor_dim0[15:0]
            g1[2] = (int)((unsigned int)Tt << 16);        // tensor_dim1[15:0]
            g1[3] = (int)(32u << 16);                     // tile_dim0 = 32
            g1[4] = 128;                                  // tile_dim1 = 128
            g1[5] = Ii;                                   // tensor_dim0_stride lo
            g1[6] = 0;
            g1[7] = 0;
            const i32x4 z4 = {0, 0, 0, 0};
            const i32x8 z8 = {0, 0, 0, 0, 0, 0, 0, 0};
            __builtin_amdgcn_tensor_load_to_lds(g0, g1, z4, z4, z8, 0);
#if __has_builtin(__builtin_amdgcn_s_wait_tensorcnt)
            __builtin_amdgcn_s_wait_tensorcnt(0);
#else
            asm volatile("s_wait_tensorcnt 0x0" ::: "memory");
#endif
        }
#else
        // fallback: manual copy (f16 already)
        #pragma unroll
        for (int cc = 0; cc < 2; ++cc) {
            const int c  = tid + cc * 256;
            const int r  = c >> 2;
            const int kc = (c & 3) * 8;
            const v8h ah = *reinterpret_cast<const v8h*>(
                hbuf + (size_t)(tBase + r) * Ii + k0 + kc);
            *reinterpret_cast<v8h*>(&As[r * LDA + kc]) = ah;
        }
#endif

        // ---------------- stage B: dequant down weights -------------
        const int   g  = k0 >> 7;
        const float sf = dsc[(size_t)g * Hh + hBase + bn];
        const int   zv = dz [(size_t)g * Hh + hBase + bn];
        #pragma unroll
        for (int j = 0; j < 4; ++j) {
            const int kr = bk0 + j * 8;
            v4h hd;
            #pragma unroll
            for (int e = 0; e < 4; ++e) {
                const size_t off = (size_t)(k0 + kr + e) * Hh + hBase + bn;
                hd[e] = (_Float16)((float)(dq[off] - zv) * sf);
            }
            *reinterpret_cast<v4h*>(&Bs[bn * LDB + kr]) = hd;
        }

        if (k0 + BK < Ii) {
            __builtin_prefetch(dq + (size_t)(k0 + BK + bk0) * Hh + hBase + bn, 0, 1);
        }

        __syncthreads();

        // ---------------- WMMA compute ------------------------------
        V16 a[4], b[2];
        #pragma unroll
        for (int mi = 0; mi < 4; ++mi) {
            const int m    = m0 + mi * 16 + l16;
            const int koff = lhalf * 8;
            a[mi].h[0] = *reinterpret_cast<const v8h*>(&As[m * LDA + koff]);
            a[mi].h[1] = *reinterpret_cast<const v8h*>(&As[m * LDA + 16 + koff]);
        }
        #pragma unroll
        for (int ni = 0; ni < 2; ++ni) {
            const int n  = n0 + ni * 16 + l16;
            const int kb = lhalf * 8;
            b[ni].h[0] = *reinterpret_cast<const v8h*>(&Bs[n * LDB + kb]);
            b[ni].h[1] = *reinterpret_cast<const v8h*>(&Bs[n * LDB + kb + 16]);
        }
        #pragma unroll
        for (int mi = 0; mi < 4; ++mi) {
            #pragma unroll
            for (int ni = 0; ni < 2; ++ni) {
                acc[mi][ni] = __builtin_amdgcn_wmma_f32_16x16x32_f16(
                    false, a[mi].v, false, b[ni].v, (short)0, acc[mi][ni], false, false);
            }
        }
        __syncthreads();
    }

    // ---------------- epilogue: f32 store ---------------------------
    #pragma unroll
    for (int mi = 0; mi < 4; ++mi) {
        #pragma unroll
        for (int ni = 0; ni < 2; ++ni) {
            #pragma unroll
            for (int r = 0; r < 8; ++r) {
                const int row = tBase + m0 + mi * 16 + lhalf * 8 + r;
                const int col = hBase + n0 + ni * 16 + l16;
                out[(size_t)row * Hh + col] = acc[mi][ni][r];
            }
        }
    }
}

extern "C" void kernel_launch(void* const* d_in, const int* in_sizes, int n_in,
                              void* d_out, int out_size, void* d_ws, size_t ws_size,
                              hipStream_t stream) {
    (void)in_sizes; (void)n_in; (void)out_size; (void)ws_size;

    const float* x   = (const float*)d_in[0];
    const int*   gq  = (const int*)  d_in[1];
    const float* gs  = (const float*)d_in[2];
    const int*   gz  = (const int*)  d_in[3];
    const int*   uq  = (const int*)  d_in[4];
    const float* us  = (const float*)d_in[5];
    const int*   uz  = (const int*)  d_in[6];
    const int*   dq  = (const int*)  d_in[7];
    const float* dsc = (const float*)d_in[8];
    const int*   dz  = (const int*)  d_in[9];
    float*       out = (float*)d_out;
    _Float16*    hb  = (_Float16*)d_ws;    // T*I f16 = 46.1 MB scratch

    dim3 gridA(Ii / BN, Tt / BM);   // 44 x 32
    gateup_swiglu_kernel<<<gridA, 256, 0, stream>>>(x, gq, gs, gz, uq, us, uz, hb);

    dim3 gridB(Hh / BN, Tt / BM);   // 16 x 32
    down_kernel<<<gridB, 256, 0, stream>>>(hb, dq, dsc, dz, out);
}
